// Net_52072183496805
// MI455X (gfx1250) — compile-verified, matrix operand order checked by reading, not run on hardware
//
#include <hip/hip_runtime.h>
#include <hip/hip_bf16.h>
#include <cstdint>

#define N_NODESC 10000
#define N_EDGESC 160000
#define E_TOTC   170000   // edges + self loops
#define F_INC    50
#define D_BODYC  1024
#define D_HEADC  256
#define NCATC    512      // [gat 256 | lin 256]
#define N_TASKC  121
#define NHC      4

typedef __bf16 bf16_t;
typedef bf16_t v16bf __attribute__((ext_vector_type(16)));
typedef float  v8f   __attribute__((ext_vector_type(8)));

union FragAB { v16bf v; uint4 q[2]; };
union FragC  { v8f v; float f[8]; };

__device__ __forceinline__ unsigned foi(float f) {  // order-preserving float->uint
  unsigned b = __float_as_uint(f);
  return (b & 0x80000000u) ? ~b : (b | 0x80000000u);
}
__device__ __forceinline__ float ifo(unsigned u) {
  unsigned b = (u & 0x80000000u) ? (u & 0x7fffffffu) : ~u;
  return __uint_as_float(b);
}
__device__ __forceinline__ float elu1(float x) { return x > 0.f ? x : (__expf(x) - 1.f); }

// ---------------------------------------------------------------------------
// Heavy per-task GEMM: hcat[M,512] = x1[M,1024] @ [Wh_t | W1_t]  (bf16 WMMA)
// Block tile 128x128, 8 waves, each wave 32x64 (2x4 WMMA 16x16x32 tiles).
// ---------------------------------------------------------------------------
__global__ __launch_bounds__(256)
void gemm_hcat_wmma(const float* __restrict__ A,   // x1 [M,1024] fp32
                    const float* __restrict__ Bh,  // Wh_t [1024,256] row-major
                    const float* __restrict__ B1,  // W1_t [1024,256] row-major
                    float* __restrict__ C,         // hcat [M,512]
                    int M)
{
  __shared__ bf16_t sA[128][32];  // A tile, bf16
  __shared__ bf16_t sB[128][32];  // B tile transposed [n][k], bf16
  const int K = D_BODYC;
  int bm = blockIdx.x * 128;
  int bn = blockIdx.y * 128;                      // global col in [0,512)
  const float* Bb = (bn < D_HEADC) ? Bh : B1;     // 128 | 256 -> tile never straddles
  int bc = (bn < D_HEADC) ? bn : bn - D_HEADC;

  int tid  = threadIdx.x;
  int lane = tid & 31;
  int wv   = tid >> 5;           // 0..7
  int wm   = (wv >> 1) * 32;     // wave M offset in tile (0,32,64,96)
  int wn   = (wv & 1) * 64;      // wave N offset in tile (0,64)
  int r0   = lane & 15;
  int ko   = (lane >> 4) * 8;    // K sub-chunk base per ISA 16-bit A layout

  v8f acc[2][4] = {};

  for (int kt = 0; kt < K; kt += 32) {
    // Stage A tile (128x32), fp32 -> bf16 (hw cvt), 4 float4 per thread
#pragma unroll
    for (int i = 0; i < 4; ++i) {
      int idx = tid * 4 + i;         // 0..1023
      int row = idx >> 3;
      int c4  = (idx & 7) << 2;
      float4 v = make_float4(0.f, 0.f, 0.f, 0.f);
      int gr = bm + row;
      if (gr < M) v = *(const float4*)(A + (size_t)gr * K + kt + c4);
      sA[row][c4 + 0] = (bf16_t)v.x;
      sA[row][c4 + 1] = (bf16_t)v.y;
      sA[row][c4 + 2] = (bf16_t)v.z;
      sA[row][c4 + 3] = (bf16_t)v.w;
    }
    // Stage B tile (32x128) transposed into sB[n][k], 4 float4 per thread
#pragma unroll
    for (int i = 0; i < 4; ++i) {
      int idx = tid * 4 + i;         // 0..1023
      int k  = idx >> 5;             // 0..31
      int nq = (idx & 31) << 2;      // 0..124
      float4 v = *(const float4*)(Bb + (size_t)(kt + k) * D_HEADC + bc + nq);
      sB[nq + 0][k] = (bf16_t)v.x;
      sB[nq + 1][k] = (bf16_t)v.y;
      sB[nq + 2][k] = (bf16_t)v.z;
      sB[nq + 3][k] = (bf16_t)v.w;
    }
    if (kt + 32 < K) {  // warm L2 for next tiles (global_prefetch_b8)
      __builtin_prefetch(Bb + (size_t)(kt + 32) * D_HEADC + bc + (tid & 127), 0, 1);
      int gr = bm + (tid >> 1);
      if (gr < M) __builtin_prefetch(A + (size_t)gr * K + kt + 32, 0, 1);
    }
    __syncthreads();

    // Fragments per ISA layout: lanes 0-15 -> rows, K chunks {ko..ko+7, ko+16..ko+23}
    FragAB a[2], b[4];
#pragma unroll
    for (int i = 0; i < 2; ++i) {
      a[i].q[0] = *(const uint4*)&sA[wm + i * 16 + r0][ko];
      a[i].q[1] = *(const uint4*)&sA[wm + i * 16 + r0][ko + 16];
    }
#pragma unroll
    for (int j = 0; j < 4; ++j) {
      b[j].q[0] = *(const uint4*)&sB[wn + j * 16 + r0][ko];
      b[j].q[1] = *(const uint4*)&sB[wn + j * 16 + r0][ko + 16];
    }
#pragma unroll
    for (int i = 0; i < 2; ++i)
#pragma unroll
      for (int j = 0; j < 4; ++j)
        acc[i][j] = __builtin_amdgcn_wmma_f32_16x16x32_bf16(
            false, a[i].v, false, b[j].v, (short)0, acc[i][j], false, false);
    __syncthreads();
  }

  // Epilogue per C/D layout: lane 0-15 -> M=e, lane 16-31 -> M=8+e; N=lane&15
  int rl = (lane >> 4) * 8;
  int cl = lane & 15;
#pragma unroll
  for (int i = 0; i < 2; ++i)
#pragma unroll
    for (int j = 0; j < 4; ++j) {
      FragC c; c.v = acc[i][j];
#pragma unroll
      for (int e = 0; e < 8; ++e) {
        int r  = bm + wm + i * 16 + rl + e;
        int cc = bn + wn + j * 16 + cl;
        if (r < M) C[(size_t)r * NCATC + cc] = c.f[e];
      }
    }
}

// ---------------------------------------------------------------------------
// Small K=50 body GEMM (1 GFLOP total): out[n,d] = A[n,:]@W[:,d] (+biases)
// ---------------------------------------------------------------------------
__global__ void gemm_small(const float* __restrict__ A, const float* __restrict__ W,
                           const float* __restrict__ bias0, const float* __restrict__ bias1,
                           float* __restrict__ out, int Nn, int K, int D)
{
  int gid = blockIdx.x * blockDim.x + threadIdx.x;
  if (gid >= Nn * D) return;
  int n = gid / D, d = gid - n * D;
  float s = 0.f;
  for (int k = 0; k < K; ++k) s += A[(size_t)n * K + k] * W[(size_t)k * D + d];
  if (bias0) s += bias0[d];
  if (bias1) s += bias1[d];
  out[gid] = s;
}

// al_src/al_dst[n,h] = sum_c h[n, h*C+c] * a[h,c]
__global__ void attn_logits(const float* __restrict__ h, int ldh,
                            const float* __restrict__ a_s, const float* __restrict__ a_d,
                            float* __restrict__ al_s, float* __restrict__ al_d,
                            int Nn, int C)
{
  int gid = blockIdx.x * blockDim.x + threadIdx.x;
  if (gid >= Nn * NHC) return;
  int n = gid >> 2, hh = gid & 3;
  const float* hp = h + (size_t)n * ldh + hh * C;
  const float* as = a_s + hh * C;
  const float* ad = a_d + hh * C;
  float s0 = 0.f, s1 = 0.f;
  for (int c = 0; c < C; ++c) { float v = hp[c]; s0 += v * as[c]; s1 += v * ad[c]; }
  al_s[gid] = s0;
  al_d[gid] = s1;
}

// segment max via ordered-uint atomicMax
__global__ void edge_max(const long long* __restrict__ srcp, const long long* __restrict__ dstp,
                         int E0, int Et,
                         const float* __restrict__ al_s, const float* __restrict__ al_d,
                         unsigned* __restrict__ m)
{
  int gid = blockIdx.x * blockDim.x + threadIdx.x;
  if (gid >= Et * NHC) return;
  int e = gid >> 2, hh = gid & 3;
  int s, d;
  if (e < E0) { s = (int)srcp[e]; d = (int)dstp[e]; } else { s = d = e - E0; }
  float ev = al_s[s * NHC + hh] + al_d[d * NHC + hh];
  ev = ev > 0.f ? ev : 0.2f * ev;              // leaky_relu(0.2)
  atomicMax(m + d * NHC + hh, foi(ev));
}

// ex = exp(e - m[dst]); denom[dst] += ex (atomic)
__global__ void edge_exp(const long long* __restrict__ srcp, const long long* __restrict__ dstp,
                         int E0, int Et,
                         const float* __restrict__ al_s, const float* __restrict__ al_d,
                         const unsigned* __restrict__ m, float* __restrict__ denom,
                         float* __restrict__ exbuf)
{
  int gid = blockIdx.x * blockDim.x + threadIdx.x;
  if (gid >= Et * NHC) return;
  int e = gid >> 2, hh = gid & 3;
  int s, d;
  if (e < E0) { s = (int)srcp[e]; d = (int)dstp[e]; } else { s = d = e - E0; }
  float ev = al_s[s * NHC + hh] + al_d[d * NHC + hh];
  ev = ev > 0.f ? ev : 0.2f * ev;
  float ex = __expf(ev - ifo(m[d * NHC + hh]));
  exbuf[e * NHC + hh] = ex;
  atomicAdd(denom + d * NHC + hh, ex);
}

// acc[dst, :] += alpha * h[src, :]  (float4 chunks, atomic adds)
__global__ void edge_scatter(const long long* __restrict__ srcp, const long long* __restrict__ dstp,
                             int E0, int Et,
                             const float* __restrict__ exbuf, const float* __restrict__ denom,
                             const float* __restrict__ hsrc, int ldh, int D, int C,
                             float* __restrict__ acc)
{
  long long gid = (long long)blockIdx.x * blockDim.x + threadIdx.x;
  int chunks = D >> 2;
  if (gid >= (long long)Et * chunks) return;
  int e = (int)(gid / chunks);
  int c = ((int)(gid - (long long)e * chunks)) << 2;
  int s, d;
  if (e < E0) { s = (int)srcp[e]; d = (int)dstp[e]; } else { s = d = e - E0; }
  int hh = c / C;
  float al = exbuf[e * NHC + hh] / (denom[d * NHC + hh] + 1e-16f);
  const float4 hv = *(const float4*)(hsrc + (size_t)s * ldh + c);
  float* ap = acc + (size_t)d * D + c;
  atomicAdd(ap + 0, al * hv.x);
  atomicAdd(ap + 1, al * hv.y);
  atomicAdd(ap + 2, al * hv.z);
  atomicAdd(ap + 3, al * hv.w);
}

__global__ void elu_kernel(const float* __restrict__ in, float* __restrict__ out, int n)
{
  int gid = blockIdx.x * blockDim.x + threadIdx.x;
  if (gid >= n) return;
  out[gid] = elu1(in[gid]);
}

// acc_h init = lin part of hcat + b1 + b_h
__global__ void init_acc_head(const float* __restrict__ hcat, const float* __restrict__ b1_t,
                              const float* __restrict__ bh_t, float* __restrict__ acc, int Nn)
{
  int gid = blockIdx.x * blockDim.x + threadIdx.x;
  if (gid >= Nn * D_HEADC) return;
  int n = gid >> 8, c = gid & 255;
  acc[gid] = hcat[(size_t)n * NCATC + D_HEADC + c] + b1_t[c] + bh_t[c];
}

// out[t,n,:] = elu(acc[n,:]) @ W2_t + b2_t  (fused ELU + 256->2)
__global__ void head_out(const float* __restrict__ acc, const float* __restrict__ W2,
                         const float* __restrict__ b2, float* __restrict__ out, int Nn)
{
  int n = blockIdx.x * blockDim.x + threadIdx.x;
  if (n >= Nn) return;
  float s0 = b2[0], s1 = b2[1];
  const float* a = acc + (size_t)n * D_HEADC;
  for (int c = 0; c < D_HEADC; ++c) {
    float v = elu1(a[c]);
    s0 += v * W2[c * 2 + 0];
    s1 += v * W2[c * 2 + 1];
  }
  out[n * 2 + 0] = s0;
  out[n * 2 + 1] = s1;
}

// ---------------------------------------------------------------------------
extern "C" void kernel_launch(void* const* d_in, const int* in_sizes, int n_in,
                              void* d_out, int out_size, void* d_ws, size_t ws_size,
                              hipStream_t stream)
{
  (void)in_sizes; (void)n_in; (void)out_size; (void)ws_size;
  const float*     x      = (const float*)d_in[0];
  const long long* ei     = (const long long*)d_in[1];   // int64 edge_index [2,E]
  const float*     W_body = (const float*)d_in[2];       // [50,4,256] == [50,1024]
  const float*     a_s_b  = (const float*)d_in[3];
  const float*     a_d_b  = (const float*)d_in[4];
  const float*     b_body = (const float*)d_in[5];
  const float*     Wfc    = (const float*)d_in[6];
  const float*     bfc    = (const float*)d_in[7];
  const float*     Wh     = (const float*)d_in[8];       // [121,1024,256]
  const float*     a_s_h  = (const float*)d_in[9];
  const float*     a_d_h  = (const float*)d_in[10];
  const float*     b_h    = (const float*)d_in[11];
  const float*     W1     = (const float*)d_in[12];      // [121,1024,256]
  const float*     b1     = (const float*)d_in[13];
  const float*     W2     = (const float*)d_in[14];      // [121,256,2]
  const float*     b2     = (const float*)d_in[15];
  float* out = (float*)d_out;

  const long long* srcp = ei;
  const long long* dstp = ei + N_EDGESC;

  // workspace carve-out (fp32 slots), ~157 MB total
  float* ws = (float*)d_ws;
  size_t off = 0;
  float* h_body = ws + off; off += (size_t)N_NODESC * D_BODYC;
  float* acc_b  = ws + off; off += (size_t)N_NODESC * D_BODYC;
  float* x1     = ws + off; off += (size_t)N_NODESC * D_BODYC;
  float* hcat   = ws + off; off += (size_t)N_NODESC * NCATC;
  float* acc_h  = ws + off; off += (size_t)N_NODESC * D_HEADC;
  float* al_s   = ws + off; off += (size_t)N_NODESC * NHC;
  float* al_d   = ws + off; off += (size_t)N_NODESC * NHC;
  unsigned* mmax = (unsigned*)(ws + off); off += (size_t)N_NODESC * NHC;  // followed by denom
  float* denom  = ws + off; off += (size_t)N_NODESC * NHC;
  float* exbuf  = ws + off; off += (size_t)E_TOTC * NHC;

  const int TPB = 256;
  const int totB = N_NODESC * D_BODYC;
  const int eThreads = E_TOTC * NHC;
  const int alThreads = N_NODESC * NHC;

  // ---------------- body ----------------
  gemm_small<<<(totB + TPB - 1) / TPB, TPB, 0, stream>>>(x, W_body, nullptr, nullptr,
                                                         h_body, N_NODESC, F_INC, D_BODYC);
  gemm_small<<<(totB + TPB - 1) / TPB, TPB, 0, stream>>>(x, Wfc, bfc, b_body,
                                                         acc_b, N_NODESC, F_INC, D_BODYC);
  attn_logits<<<(alThreads + TPB - 1) / TPB, TPB, 0, stream>>>(h_body, D_BODYC, a_s_b, a_d_b,
                                                               al_s, al_d, N_NODESC, 256);
  hipMemsetAsync(mmax, 0, (size_t)N_NODESC * NHC * 2 * sizeof(float), stream); // m + denom
  edge_max<<<(eThreads + TPB - 1) / TPB, TPB, 0, stream>>>(srcp, dstp, N_EDGESC, E_TOTC,
                                                           al_s, al_d, mmax);
  edge_exp<<<(eThreads + TPB - 1) / TPB, TPB, 0, stream>>>(srcp, dstp, N_EDGESC, E_TOTC,
                                                           al_s, al_d, mmax, denom, exbuf);
  long long scatB = (long long)E_TOTC * (D_BODYC / 4);
  edge_scatter<<<(unsigned)((scatB + TPB - 1) / TPB), TPB, 0, stream>>>(
      srcp, dstp, N_EDGESC, E_TOTC, exbuf, denom, h_body, D_BODYC, D_BODYC, 256, acc_b);
  elu_kernel<<<(totB + TPB - 1) / TPB, TPB, 0, stream>>>(acc_b, x1, totB);

  // ---------------- 121 task heads ----------------
  dim3 gGemm((N_NODESC + 127) / 128, NCATC / 128);
  long long scatH = (long long)E_TOTC * (D_HEADC / 4);
  const int totH = N_NODESC * D_HEADC;
  for (int t = 0; t < N_TASKC; ++t) {
    const float* Wh_t = Wh + (size_t)t * D_BODYC * D_HEADC;
    const float* W1_t = W1 + (size_t)t * D_BODYC * D_HEADC;
    const float* as_t = a_s_h + (size_t)t * D_HEADC;
    const float* ad_t = a_d_h + (size_t)t * D_HEADC;
    const float* bh_t = b_h + (size_t)t * D_HEADC;
    const float* b1_t = b1 + (size_t)t * D_HEADC;
    const float* W2_t = W2 + (size_t)t * D_HEADC * 2;
    const float* b2_t = b2 + (size_t)t * 2;
    float* out_t = out + (size_t)t * N_NODESC * 2;

    gemm_hcat_wmma<<<gGemm, TPB, 0, stream>>>(x1, Wh_t, W1_t, hcat, N_NODESC);
    attn_logits<<<(alThreads + TPB - 1) / TPB, TPB, 0, stream>>>(hcat, NCATC, as_t, ad_t,
                                                                 al_s, al_d, N_NODESC, 64);
    hipMemsetAsync(mmax, 0, (size_t)N_NODESC * NHC * 2 * sizeof(float), stream);
    edge_max<<<(eThreads + TPB - 1) / TPB, TPB, 0, stream>>>(srcp, dstp, N_EDGESC, E_TOTC,
                                                             al_s, al_d, mmax);
    edge_exp<<<(eThreads + TPB - 1) / TPB, TPB, 0, stream>>>(srcp, dstp, N_EDGESC, E_TOTC,
                                                             al_s, al_d, mmax, denom, exbuf);
    init_acc_head<<<(totH + TPB - 1) / TPB, TPB, 0, stream>>>(hcat, b1_t, bh_t, acc_h, N_NODESC);
    edge_scatter<<<(unsigned)((scatH + TPB - 1) / TPB), TPB, 0, stream>>>(
        srcp, dstp, N_EDGESC, E_TOTC, exbuf, denom, hcat, NCATC, D_HEADC, 64, acc_h);
    head_out<<<(N_NODESC + 127) / 128, 128, 0, stream>>>(acc_h, W2_t, b2_t, out_t, N_NODESC);
  }
}
